// HybridGCNGAT_79809082294375
// MI455X (gfx1250) — matrix operand order repaired and checked by reading, not compiled, not run on hardware
//
#include <hip/hip_runtime.h>
#include <hip/hip_bf16.h>

#define HID 128
#define NHEAD 4
#define CHEAD 32
#define NGRAPH 64
#define NOUT 10
#define LN_EPS 1e-5f

typedef __attribute__((ext_vector_type(16))) __bf16 v16bf;
typedef __attribute__((ext_vector_type(8)))  float v8f;

union Frag16 { v16bf v; unsigned int w[8]; };

__device__ __forceinline__ float lrelu(float v) { return v >= 0.0f ? v : 0.2f * v; }

// round-to-nearest-even f32 -> bf16 bits (avoids relying on scalar __bf16 conversions)
__device__ __forceinline__ unsigned short f2bf(float f) {
    unsigned int u = __float_as_uint(f);
    unsigned int r = (u + 0x7FFFu + ((u >> 16) & 1u)) >> 16;
    return (unsigned short)r;
}

__device__ __forceinline__ void atomicMaxF(float* addr, float val) {
    if (val >= 0.0f) atomicMax((int*)addr, __float_as_int(val));
    else             atomicMin((unsigned int*)addr, __float_as_uint(val));
}

// ---------------------------------------------------------------------------
// generic fills / conversions
// ---------------------------------------------------------------------------
__global__ void k_fill(float* p, float v, int n) {
    int i = blockIdx.x * blockDim.x + threadIdx.x;
    if (i < n) p[i] = v;
}

__global__ void k_f32_to_bf16(const float* __restrict__ src, unsigned short* __restrict__ dst, int n) {
    int i = blockIdx.x * blockDim.x + threadIdx.x;
    if (i < n) dst[i] = f2bf(src[i]);
}

// ---------------------------------------------------------------------------
// WMMA GEMM: out[N,128] = A[N,128](bf16) @ W[128,128](f32->bf16 in LDS)
// mode 0: plain (outF only)
// mode 1: input projection: leaky(acc + bias) then eval-BatchNorm, outF + outBF
// Each block: 128 threads = 4 waves; wave w computes rows [base+16w, base+16w+16)
// ---------------------------------------------------------------------------
__global__ __launch_bounds__(128) void k_gemm_node(
    const unsigned short* __restrict__ A, const float* __restrict__ W,
    const float* __restrict__ bias,
    const float* __restrict__ bng, const float* __restrict__ bnb,
    const float* __restrict__ bnrm, const float* __restrict__ bnrv,
    float* __restrict__ outF, unsigned short* __restrict__ outBF,
    int nrows, int mode)
{
    __shared__ unsigned short wlds[HID * HID];   // transposed: wlds[n*128 + k] (32 KB)

    for (int i = threadIdx.x; i < HID * HID; i += blockDim.x) {
        int k = i >> 7, n = i & 127;
        wlds[n * HID + k] = f2bf(W[i]);          // W row-major [k, n]
    }
    __syncthreads();

    const int wave    = threadIdx.x >> 5;
    const int lane    = threadIdx.x & 31;
    const int halfsel = lane >> 4;               // 0: lanes 0-15, 1: lanes 16-31
    const int l16     = lane & 15;
    const int rowbase = blockIdx.x * 64 + wave * 16;

    v8f acc[8];
#pragma unroll
    for (int t = 0; t < 8; ++t) acc[t] = v8f{0.f,0.f,0.f,0.f,0.f,0.f,0.f,0.f};

    int arow = rowbase + l16;
    if (arow >= nrows) arow = nrows - 1;         // clamp; stores are guarded
    const unsigned int* arp = (const unsigned int*)(A + (size_t)arow * HID);

#pragma unroll
    for (int ks = 0; ks < 4; ++ks) {             // K = 128 in steps of 32
        Frag16 af;
#pragma unroll
        for (int v = 0; v < 8; ++v) {
            // 16-bit A fragment layout: lanes<16 hold K {0..7,16..23}, lanes>=16 hold K {8..15,24..31}
            int k = ks * 32 + halfsel * 8 + (v & 3) * 2 + (v >> 2) * 16;
            af.w[v] = arp[k >> 1];               // packed bf16 pair
        }
#pragma unroll
        for (int t = 0; t < 8; ++t) {            // 8 column tiles of 16
            Frag16 bf_;
            int n = t * 16 + l16;
            const unsigned int* bp = (const unsigned int*)(wlds + n * HID);
#pragma unroll
            for (int v = 0; v < 8; ++v) {
                // 16-bit B fragment layout: lanes<16 hold K {0..7,16..23}, lanes>=16 hold K {8..15,24..31}
                int k = ks * 32 + halfsel * 8 + (v & 3) * 2 + (v >> 2) * 16;
                bf_.w[v] = bp[k >> 1];
            }
            acc[t] = __builtin_amdgcn_wmma_f32_16x16x32_bf16(
                false, af.v, false, bf_.v, (short)0, acc[t], false, false);
        }
    }

#pragma unroll
    for (int t = 0; t < 8; ++t) {
        int col = t * 16 + l16;
#pragma unroll
        for (int r = 0; r < 8; ++r) {
            int row = rowbase + halfsel * 8 + r;  // C/D layout: vgpr r -> M = r + 8*half
            if (row >= nrows) continue;
            float v = acc[t][r];
            if (mode == 1) {
                v += bias[col];
                v = lrelu(v);
                v = (v - bnrm[col]) * rsqrtf(bnrv[col] + LN_EPS) * bng[col] + bnb[col];
            }
            outF[(size_t)row * HID + col] = v;
            if (outBF) outBF[(size_t)row * HID + col] = f2bf(v);
        }
    }
}

// ---------------------------------------------------------------------------
// degree / normalization
// ---------------------------------------------------------------------------
__global__ void k_deg(const int* __restrict__ ei, float* __restrict__ deg, int E) {
    int e = blockIdx.x * blockDim.x + threadIdx.x;
    if (e < E) atomicAdd(&deg[ei[E + e]], 1.0f);
}
__global__ void k_rsqrt_inplace(float* p, int n) {
    int i = blockIdx.x * blockDim.x + threadIdx.x;
    if (i < n) p[i] = rsqrtf(fmaxf(p[i], 1.0f));
}

// ---------------------------------------------------------------------------
// GCN coefficient scatter over extended edge list (E real edges + N self loops)
// thread handles (edge, 4 channels)
// ---------------------------------------------------------------------------
__global__ void k_gcn_scatter(const int* __restrict__ ei, const float* __restrict__ dis,
                              const float* __restrict__ vw, float* __restrict__ agg,
                              int E, int N)
{
    int gid = blockIdx.x * blockDim.x + threadIdx.x;
    int total = (E + N) * 32;
    if (gid >= total) return;
    int e = gid >> 5, c0 = (gid & 31) * 4;
    int s, d;
    if (e < E) { s = ei[e]; d = ei[E + e]; } else { s = d = e - E; }
    float coef = dis[s] * dis[d];
    const float4 v = *(const float4*)(vw + (size_t)s * HID + c0);
    float* o = agg + (size_t)d * HID + c0;
    atomicAdd(o + 0, coef * v.x);
    atomicAdd(o + 1, coef * v.y);
    atomicAdd(o + 2, coef * v.z);
    atomicAdd(o + 3, coef * v.w);
}

// ---------------------------------------------------------------------------
// bias + LayerNorm + LeakyReLU, one wave32 per node, shuffle reductions
// ---------------------------------------------------------------------------
__global__ void k_ln(const float* __restrict__ agg, const float* __restrict__ bias,
                     const float* __restrict__ g, const float* __restrict__ b,
                     float* __restrict__ outF, unsigned short* __restrict__ outBF, int N)
{
    int node = blockIdx.x * 8 + (threadIdx.x >> 5);
    if (node >= N) return;
    int lane = threadIdx.x & 31;
    int c0 = lane * 4;
    float4 v = *(const float4*)(agg + (size_t)node * HID + c0);
    v.x += bias[c0 + 0]; v.y += bias[c0 + 1]; v.z += bias[c0 + 2]; v.w += bias[c0 + 3];

    float s = v.x + v.y + v.z + v.w;
#pragma unroll
    for (int off = 16; off > 0; off >>= 1) s += __shfl_xor(s, off, 32);
    float mu = s * (1.0f / HID);

    float dx = v.x - mu, dy = v.y - mu, dz = v.z - mu, dw = v.w - mu;
    float q = dx * dx + dy * dy + dz * dz + dw * dw;
#pragma unroll
    for (int off = 16; off > 0; off >>= 1) q += __shfl_xor(q, off, 32);
    float rs = rsqrtf(q * (1.0f / HID) + LN_EPS);

    float o[4] = { dx, dy, dz, dw };
#pragma unroll
    for (int j = 0; j < 4; ++j) {
        int c = c0 + j;
        float y = lrelu(o[j] * rs * g[c] + b[c]);
        outF[(size_t)node * HID + c] = y;
        if (outBF) outBF[(size_t)node * HID + c] = f2bf(y);
    }
}

// ---------------------------------------------------------------------------
// GATv2: logits per (edge, head) + segment max ; one wave per extended edge
// ---------------------------------------------------------------------------
__global__ void k_gat_logits(const int* __restrict__ ei,
                             const float* __restrict__ xl, const float* __restrict__ xr,
                             const float* __restrict__ att,
                             float* __restrict__ logits, float* __restrict__ maxv,
                             int E, int N)
{
    int e = blockIdx.x * 8 + (threadIdx.x >> 5);
    if (e >= E + N) return;
    int lane = threadIdx.x & 31;
    int s, d;
    if (e < E) { s = ei[e]; d = ei[E + e]; } else { s = d = e - E; }
#pragma unroll
    for (int h = 0; h < NHEAD; ++h) {
        int c = h * CHEAD + lane;
        float t = lrelu(xl[(size_t)s * HID + c] + xr[(size_t)d * HID + c]) * att[h * CHEAD + lane];
#pragma unroll
        for (int off = 16; off > 0; off >>= 1) t += __shfl_xor(t, off, 32);
        if (lane == 0) {
            logits[(size_t)e * NHEAD + h] = t;
            atomicMaxF(&maxv[(size_t)d * NHEAD + h], t);
        }
    }
}

__global__ void k_gat_exp(const int* __restrict__ ei, const float* __restrict__ logits,
                          const float* __restrict__ maxv,
                          float* __restrict__ alpha, float* __restrict__ denom,
                          int E, int N)
{
    int gid = blockIdx.x * blockDim.x + threadIdx.x;
    int total = (E + N) * NHEAD;
    if (gid >= total) return;
    int e = gid >> 2, h = gid & 3;
    int d = (e < E) ? ei[E + e] : (e - E);
    float a = __expf(logits[gid] - maxv[(size_t)d * NHEAD + h]);
    alpha[gid] = a;
    atomicAdd(&denom[(size_t)d * NHEAD + h], a);
}

__global__ void k_gat_scatter(const int* __restrict__ ei,
                              const float* __restrict__ alpha, const float* __restrict__ denom,
                              const float* __restrict__ xl, float* __restrict__ agg,
                              int E, int N)
{
    int gid = blockIdx.x * blockDim.x + threadIdx.x;
    int total = (E + N) * 32;
    if (gid >= total) return;
    int e = gid >> 5, r = gid & 31;
    int h = r >> 3, c0 = h * CHEAD + (r & 7) * 4;
    int s, d;
    if (e < E) { s = ei[e]; d = ei[E + e]; } else { s = d = e - E; }
    float dn = denom[(size_t)d * NHEAD + h];
    float w = alpha[(size_t)e * NHEAD + h] / fmaxf(dn, 1e-30f);
    const float4 v = *(const float4*)(xl + (size_t)s * HID + c0);
    float* o = agg + (size_t)d * HID + c0;
    atomicAdd(o + 0, w * v.x);
    atomicAdd(o + 1, w * v.y);
    atomicAdd(o + 2, w * v.z);
    atomicAdd(o + 3, w * v.w);
}

// ---------------------------------------------------------------------------
// global mean pool: atomic sum into pooled[G,128] + counts
// ---------------------------------------------------------------------------
__global__ void k_pool(const int* __restrict__ batch, const float* __restrict__ h,
                       float* __restrict__ pooled, float* __restrict__ counts, int N)
{
    int gid = blockIdx.x * blockDim.x + threadIdx.x;
    if (gid >= N * 32) return;
    int n = gid >> 5, lane = gid & 31, c0 = lane * 4;
    int g = batch[n];
    const float4 v = *(const float4*)(h + (size_t)n * HID + c0);
    float* o = pooled + (size_t)g * HID + c0;
    atomicAdd(o + 0, v.x);
    atomicAdd(o + 1, v.y);
    atomicAdd(o + 2, v.z);
    atomicAdd(o + 3, v.w);
    if (lane == 0) atomicAdd(&counts[g], 1.0f);
}

// ---------------------------------------------------------------------------
// MLP head (tiny: [64,128]@[128,64] -> bn -> [64,64]@[64,10]); single block
// ---------------------------------------------------------------------------
__global__ void k_head(const float* __restrict__ pooled, const float* __restrict__ counts,
                       const float* __restrict__ W1, const float* __restrict__ b1,
                       const float* __restrict__ bng, const float* __restrict__ bnb,
                       const float* __restrict__ bnrm, const float* __restrict__ bnrv,
                       const float* __restrict__ W2, const float* __restrict__ b2,
                       float* __restrict__ out)
{
    __shared__ float t1[NGRAPH * 64];
    for (int i = threadIdx.x; i < NGRAPH * 64; i += blockDim.x) {
        int g = i >> 6, j = i & 63;
        float inv = 1.0f / fmaxf(counts[g], 1.0f);
        float acc = 0.0f;
        for (int k = 0; k < HID; ++k) acc += pooled[g * HID + k] * inv * W1[k * 64 + j];
        acc = lrelu(acc + b1[j]);
        acc = (acc - bnrm[j]) * rsqrtf(bnrv[j] + LN_EPS) * bng[j] + bnb[j];
        t1[i] = acc;
    }
    __syncthreads();
    for (int i = threadIdx.x; i < NGRAPH * NOUT; i += blockDim.x) {
        int g = i / NOUT, o = i % NOUT;
        float acc = b2[o];
        for (int k = 0; k < 64; ++k) acc += t1[g * 64 + k] * W2[k * NOUT + o];
        out[i] = acc;
    }
}

// ---------------------------------------------------------------------------
// host launcher
// ---------------------------------------------------------------------------
static inline int cdiv(long long a, int b) { return (int)((a + b - 1) / b); }

extern "C" void kernel_launch(void* const* d_in, const int* in_sizes, int n_in,
                              void* d_out, int out_size, void* d_ws, size_t ws_size,
                              hipStream_t stream)
{
    const float* x     = (const float*)d_in[0];
    const int*   ei    = (const int*)d_in[1];
    const int*   batch = (const int*)d_in[2];

    const float* in_W   = (const float*)d_in[3];
    const float* in_b   = (const float*)d_in[4];
    const float* in_bng = (const float*)d_in[5];
    const float* in_bnb = (const float*)d_in[6];
    const float* in_rm  = (const float*)d_in[7];
    const float* in_rv  = (const float*)d_in[8];
    const float* g0_W   = (const float*)d_in[9];
    const float* g0_b   = (const float*)d_in[10];
    const float* ln0_g  = (const float*)d_in[11];
    const float* ln0_b  = (const float*)d_in[12];
    const float* g1_W   = (const float*)d_in[13];
    const float* g1_b   = (const float*)d_in[14];
    const float* ln1_g  = (const float*)d_in[15];
    const float* ln1_b  = (const float*)d_in[16];
    const float* gat_Wl = (const float*)d_in[17];
    const float* gat_Wr = (const float*)d_in[18];
    const float* gat_at = (const float*)d_in[19];
    const float* gat_b  = (const float*)d_in[20];
    const float* ln2_g  = (const float*)d_in[21];
    const float* ln2_b  = (const float*)d_in[22];
    const float* h_W1   = (const float*)d_in[23];
    const float* h_b1   = (const float*)d_in[24];
    const float* h_bng  = (const float*)d_in[25];
    const float* h_bnb  = (const float*)d_in[26];
    const float* h_rm   = (const float*)d_in[27];
    const float* h_rv   = (const float*)d_in[28];
    const float* h_W2   = (const float*)d_in[29];
    const float* h_b2   = (const float*)d_in[30];

    const int N  = in_sizes[0] / HID;
    const int E  = in_sizes[1] / 2;
    const int EX = E + N;

    // carve workspace (256B aligned)
    char* ws = (char*)d_ws;
    auto carve = [&](size_t bytes) {
        void* p = (void*)ws;
        ws += (bytes + 255) & ~(size_t)255;
        return p;
    };
    unsigned short* xbf  = (unsigned short*)carve((size_t)N * HID * 2);
    unsigned short* hbf  = (unsigned short*)carve((size_t)N * HID * 2);
    float* hF    = (float*)carve((size_t)N * HID * 4);
    float* vw    = (float*)carve((size_t)N * HID * 4);   // also GAT xl
    float* xr    = (float*)carve((size_t)N * HID * 4);
    float* agg   = (float*)carve((size_t)N * HID * 4);
    float* dis   = (float*)carve((size_t)N * 4);
    float* logit = (float*)carve((size_t)EX * NHEAD * 4);
    float* alpha = (float*)carve((size_t)EX * NHEAD * 4);
    float* maxv  = (float*)carve((size_t)N * NHEAD * 4);
    float* denom = (float*)carve((size_t)N * NHEAD * 4);
    float* pooled = (float*)carve((size_t)NGRAPH * HID * 4);
    float* counts = (float*)carve((size_t)NGRAPH * 4);

    const int B = 256;
    const int gemm_grid = cdiv(N, 64);

    // input projection: h = BN(leaky(x @ in_W + in_b))
    k_f32_to_bf16<<<cdiv((long long)N * HID, B), B, 0, stream>>>(x, xbf, N * HID);
    k_gemm_node<<<gemm_grid, 128, 0, stream>>>(xbf, in_W, in_b, in_bng, in_bnb, in_rm, in_rv,
                                               hF, hbf, N, 1);

    // symmetric GCN normalization (self loops => deg starts at 1)
    k_fill<<<cdiv(N, B), B, 0, stream>>>(dis, 1.0f, N);
    k_deg<<<cdiv(E, B), B, 0, stream>>>(ei, dis, E);
    k_rsqrt_inplace<<<cdiv(N, B), B, 0, stream>>>(dis, N);

    // GCN layer 0
    k_gemm_node<<<gemm_grid, 128, 0, stream>>>(hbf, g0_W, nullptr, nullptr, nullptr, nullptr,
                                               nullptr, vw, nullptr, N, 0);
    k_fill<<<cdiv((long long)N * HID, B), B, 0, stream>>>(agg, 0.0f, N * HID);
    k_gcn_scatter<<<cdiv((long long)EX * 32, B), B, 0, stream>>>(ei, dis, vw, agg, E, N);
    k_ln<<<cdiv(N, 8), 256, 0, stream>>>(agg, g0_b, ln0_g, ln0_b, hF, hbf, N);

    // GCN layer 1
    k_gemm_node<<<gemm_grid, 128, 0, stream>>>(hbf, g1_W, nullptr, nullptr, nullptr, nullptr,
                                               nullptr, vw, nullptr, N, 0);
    k_fill<<<cdiv((long long)N * HID, B), B, 0, stream>>>(agg, 0.0f, N * HID);
    k_gcn_scatter<<<cdiv((long long)EX * 32, B), B, 0, stream>>>(ei, dis, vw, agg, E, N);
    k_ln<<<cdiv(N, 8), 256, 0, stream>>>(agg, g1_b, ln1_g, ln1_b, hF, hbf, N);

    // GATv2 layer
    k_gemm_node<<<gemm_grid, 128, 0, stream>>>(hbf, gat_Wl, nullptr, nullptr, nullptr, nullptr,
                                               nullptr, vw, nullptr, N, 0);   // xl
    k_gemm_node<<<gemm_grid, 128, 0, stream>>>(hbf, gat_Wr, nullptr, nullptr, nullptr, nullptr,
                                               nullptr, xr, nullptr, N, 0);
    k_fill<<<cdiv(N * NHEAD, B), B, 0, stream>>>(maxv, -3.0e38f, N * NHEAD);
    k_fill<<<cdiv(N * NHEAD, B), B, 0, stream>>>(denom, 0.0f, N * NHEAD);
    k_gat_logits<<<cdiv(EX, 8), 256, 0, stream>>>(ei, vw, xr, gat_at, logit, maxv, E, N);
    k_gat_exp<<<cdiv((long long)EX * NHEAD, B), B, 0, stream>>>(ei, logit, maxv, alpha, denom, E, N);
    k_fill<<<cdiv((long long)N * HID, B), B, 0, stream>>>(agg, 0.0f, N * HID);
    k_gat_scatter<<<cdiv((long long)EX * 32, B), B, 0, stream>>>(ei, alpha, denom, vw, agg, E, N);
    k_ln<<<cdiv(N, 8), 256, 0, stream>>>(agg, gat_b, ln2_g, ln2_b, hF, nullptr, N);

    // global mean pool + head
    k_fill<<<cdiv(NGRAPH * HID, B), B, 0, stream>>>(pooled, 0.0f, NGRAPH * HID);
    k_fill<<<1, NGRAPH, 0, stream>>>(counts, 0.0f, NGRAPH);
    k_pool<<<cdiv((long long)N * 32, B), B, 0, stream>>>(batch, hF, pooled, counts, N);
    k_head<<<1, 256, 0, stream>>>(pooled, counts, h_W1, h_b1, h_bng, h_bnb, h_rm, h_rv,
                                  h_W2, h_b2, (float*)d_out);
}
